// SkipGramModel_33895881900158
// MI455X (gfx1250) — matrix operand order repaired and checked by reading, not compiled, not run on hardware
//
#include <hip/hip_runtime.h>
#include <hip/hip_bf16.h>

typedef __attribute__((ext_vector_type(16))) __bf16 v16bf;
typedef __attribute__((ext_vector_type(8)))  float  v8f;

#define DIM  128
#define KNEG 16

__device__ __forceinline__ float log_sigmoid(float x) {
    // numerically stable: min(x,0) - log1p(exp(-|x|))
    return fminf(x, 0.0f) - log1pf(expf(-fabsf(x)));
}

__global__ void __launch_bounds__(256)
skipgram_loss_kernel(const int* __restrict__ centrals,
                     const int* __restrict__ pos_ctx,
                     const int* __restrict__ neg_ctx,
                     const float* __restrict__ word_emb,
                     const float* __restrict__ con_emb,
                     float* __restrict__ block_partials,
                     int B)
{
    const int lane            = threadIdx.x & 31;
    const int wave_in_block   = threadIdx.x >> 5;
    const int waves_per_block = blockDim.x >> 5;
    const int wave_id         = blockIdx.x * waves_per_block + wave_in_block;
    const int total_waves     = gridDim.x * waves_per_block;

    const int half = lane >> 4;   // 0: lanes 0-15, 1: lanes 16-31
    const int row  = lane & 15;   // WMMA matrix row handled by this lane

    float wave_acc = 0.0f;

    // Loop condition is wave-uniform (b uniform across lanes) -> EXEC stays all-ones
    for (int b = wave_id; b < B; b += total_waves) {
        const float* wrow = word_emb + (size_t)centrals[b] * DIM;
        const float* crow = con_emb  + (size_t)pos_ctx[b]  * DIM;
        const float* nrow = con_emb  + (size_t)neg_ctx[b * KNEG + row] * DIM;

        // Prefetch next iteration's gather rows into L2/L0 (global_prefetch_b8)
        {
            int bn = b + total_waves;
            if (bn < B) {
                const float* wn = word_emb + (size_t)centrals[bn] * DIM;
                const float* cn = con_emb  + (size_t)pos_ctx[bn]  * DIM;
                const float* nn = con_emb  + (size_t)neg_ctx[bn * KNEG + row] * DIM;
                __builtin_prefetch(wn + lane * 4, 0, 3);
                __builtin_prefetch(cn + lane * 4, 0, 3);
                __builtin_prefetch(nn + half * 64, 0, 3);  // half a row per lane-half
            }
        }

        // ---------- positive score: exact f32 dot + butterfly reduce ----------
        float4 wv = *(const float4*)(wrow + lane * 4);
        float4 cv = *(const float4*)(crow + lane * 4);
        float p = wv.x * cv.x + wv.y * cv.y + wv.z * cv.z + wv.w * cv.w;
        #pragma unroll
        for (int m = 16; m >= 1; m >>= 1) p += __shfl_xor(p, m, 32);
        // p == pos_score in every lane

        // ---------- negative scores: bf16 WMMA, w broadcast into all B columns ----------
        v8f acc = {};
        #pragma unroll
        for (int kc = 0; kc < DIM / 32; ++kc) {
            const int dbase = kc * 32;

            // A-matrix (16x32 bf16): lane layout per CDNA5 ISA 7.12.2
            const float* ap = nrow + dbase + half * 8;
            float4 a0 = *(const float4*)(ap + 0);
            float4 a1 = *(const float4*)(ap + 4);
            float4 a2 = *(const float4*)(ap + 16);
            float4 a3 = *(const float4*)(ap + 20);

            // B-matrix (32x16 bf16): every column = w chunk; lane needs K = half*8 + 0..7, 16 + half*8 + 0..7
            const float* bp = wrow + dbase + half * 8;
            float4 b0 = *(const float4*)(bp + 0);
            float4 b1 = *(const float4*)(bp + 4);
            float4 b2 = *(const float4*)(bp + 16);
            float4 b3 = *(const float4*)(bp + 20);

            v16bf A, Bm;
            A[0]  = (__bf16)a0.x; A[1]  = (__bf16)a0.y; A[2]  = (__bf16)a0.z; A[3]  = (__bf16)a0.w;
            A[4]  = (__bf16)a1.x; A[5]  = (__bf16)a1.y; A[6]  = (__bf16)a1.z; A[7]  = (__bf16)a1.w;
            A[8]  = (__bf16)a2.x; A[9]  = (__bf16)a2.y; A[10] = (__bf16)a2.z; A[11] = (__bf16)a2.w;
            A[12] = (__bf16)a3.x; A[13] = (__bf16)a3.y; A[14] = (__bf16)a3.z; A[15] = (__bf16)a3.w;

            Bm[0]  = (__bf16)b0.x; Bm[1]  = (__bf16)b0.y; Bm[2]  = (__bf16)b0.z; Bm[3]  = (__bf16)b0.w;
            Bm[4]  = (__bf16)b1.x; Bm[5]  = (__bf16)b1.y; Bm[6]  = (__bf16)b1.z; Bm[7]  = (__bf16)b1.w;
            Bm[8]  = (__bf16)b2.x; Bm[9]  = (__bf16)b2.y; Bm[10] = (__bf16)b2.z; Bm[11] = (__bf16)b2.w;
            Bm[12] = (__bf16)b3.x; Bm[13] = (__bf16)b3.y; Bm[14] = (__bf16)b3.z; Bm[15] = (__bf16)b3.w;

            acc = __builtin_amdgcn_wmma_f32_16x16x32_bf16(
                false, A, false, Bm, (short)0, acc, false, false);
        }

        // D layout: lanes 0-15 hold rows 0-7 (neg k=0..7), lanes 16-31 rows 8-15
        float nl = 0.0f;
        #pragma unroll
        for (int r = 0; r < 8; ++r) nl += log_sigmoid(-acc[r]);
        nl += __shfl_xor(nl, 16, 32);   // total neg_loss in all lanes

        wave_acc += log_sigmoid(p) + nl;
    }

    // ---------- deterministic block reduction ----------
    __shared__ float smem[32];
    if (lane == 0) smem[wave_in_block] = wave_acc;
    __syncthreads();
    if (threadIdx.x == 0) {
        float s = 0.0f;
        for (int i = 0; i < waves_per_block; ++i) s += smem[i];
        block_partials[blockIdx.x] = s;
    }
}

__global__ void __launch_bounds__(256)
skipgram_reduce_kernel(const float* __restrict__ partials, int n,
                       float* __restrict__ out, float invB)
{
    __shared__ float s[256];
    float acc = 0.0f;
    for (int i = threadIdx.x; i < n; i += 256) acc += partials[i];
    s[threadIdx.x] = acc;
    __syncthreads();
    for (int off = 128; off > 0; off >>= 1) {
        if (threadIdx.x < off) s[threadIdx.x] += s[threadIdx.x + off];
        __syncthreads();
    }
    if (threadIdx.x == 0) out[0] = -s[0] * invB;
}

extern "C" void kernel_launch(void* const* d_in, const int* in_sizes, int n_in,
                              void* d_out, int out_size, void* d_ws, size_t ws_size,
                              hipStream_t stream) {
    const int*   centrals = (const int*)d_in[0];   // [B]
    const int*   pos_ctx  = (const int*)d_in[1];   // [B]
    const int*   neg_ctx  = (const int*)d_in[2];   // [B, 16]
    const float* word_emb = (const float*)d_in[3]; // [V, 128]
    const float* con_emb  = (const float*)d_in[4]; // [V, 128]
    float*       out      = (float*)d_out;
    const int    B        = in_sizes[0];

    int blocks = 1024;                              // 8192 waves, 4 batch elems each
    if ((size_t)blocks * sizeof(float) > ws_size)   // safety clamp for scratch
        blocks = (int)(ws_size / sizeof(float));
    if (blocks < 1) blocks = 1;
    float* partials = (float*)d_ws;

    skipgram_loss_kernel<<<blocks, 256, 0, stream>>>(
        centrals, pos_ctx, neg_ctx, word_emb, con_emb, partials, B);
    skipgram_reduce_kernel<<<1, 256, 0, stream>>>(
        partials, blocks, out, 1.0f / (float)B);
}
